// InteractionBlock_14714557956330
// MI455X (gfx1250) — matrix operand order repaired and checked by reading, not compiled, not run on hardware
//
#include <hip/hip_runtime.h>
#include <hip/hip_bf16.h>

typedef __attribute__((ext_vector_type(16))) _Float16 v16h;
typedef __attribute__((ext_vector_type(8)))  float    v8f;

#define DEV __device__ __forceinline__

#define INV2f  0.70710678118654752f
#define INV3f  0.57735026918962576f
#define INVNBf 0.5f

// silu with native-rate rcp (v_rcp_f32) instead of IEEE division sequence.
DEV float silu(float x) { return x * __builtin_amdgcn_rcpf(1.0f + __expf(-x)); }

// ---- WMMA helpers (gfx1250, wave32, 16x16x32 f16 -> f32) -------------------
// A tile: 16 rows x 32 K, row-major f16 in LDS (leading dim = lda halfs).
// Per ISA 7.12.2: lanes 0-15 -> M=lane, K in {0..7, 16..23};
//                 lanes 16-31 -> M=lane-16, K in {8..15, 24..31}.
// Each lane reads two contiguous 16B runs -> 2x ds_load_b128.
DEV v16h load_a16(const _Float16* a, int lda, int lane) {
  const int row = lane & 15;
  const int k0  = (lane < 16) ? 0 : 8;
  const _Float16* p = a + row * lda + k0;
  v16h r;
#pragma unroll
  for (int i = 0; i < 8; ++i) r[i] = p[i];
#pragma unroll
  for (int i = 0; i < 8; ++i) r[i + 8] = p[16 + i];
  return r;
}

// B fragments are pre-swizzled in LDS: per 32x16 tile, lane's 16 halfs are
// contiguous -> a single 32B run per lane (2x ds_load_b128).
DEV v16h load_bswz(const _Float16* wswz, int tile, int lane) {
  const _Float16* p = wswz + ((tile << 5) + lane) * 16;
  v16h r;
#pragma unroll
  for (int i = 0; i < 16; ++i) r[i] = p[i];
  return r;
}

DEV v8f wmma_f16(v16h a, v16h b, v8f c) {
  return __builtin_amdgcn_wmma_f32_16x16x32_f16(false, a, false, b, (short)0, c,
                                                false, false);
}

// ---- weight staging: f32 row-major global -> f16 B-fragment-swizzled LDS ---
// dst holds (K/32)*(ncols/16) tiles of 512 halfs, tile t = kc*(ncols/16)+jn.
// Within a tile: [lane][i] = B[kc*32 + (lane<16?0:8) + i + (i>=8?8:0)][jn*16 + (lane&15)].
// Rows >= realK are zero-padded (for K=16 weights padded to K=32).
DEV void stage_w_swz(_Float16* dst, const float* src, int K, int ncols,
                     int realK, int tid, int nt) {
  const int ntj = ncols >> 4;
  const int total = K * ncols;
  for (int idx = tid; idx < total; idx += nt) {
    int t    = idx >> 9;
    int rem  = idx & 511;
    int lane = rem >> 4;
    int i    = rem & 15;
    int kc = t / ntj, jn = t - kc * ntj;
    int col = lane & 15;
    int k0  = (lane < 16) ? 0 : 8;
    int k   = kc * 32 + k0 + i + ((i >= 8) ? 8 : 0);
    int n   = jn * 16 + col;
    dst[idx] = (k < realK) ? (_Float16)src[k * ncols + n] : (_Float16)0.0f;
  }
}

// ---- residual block on a 16x64 tile:  x <- (x + silu(silu(x@W1)@W2))/sqrt2
// W1/W2: 64x64 swizzled (8 tiles, tile = c*4 + j).
// xbuf: f32 skip copy; aIn: f16 copy of x (A source); aTmp: scratch f16 tile.
DEV void residual_tile(float* xbuf, _Float16* aIn, _Float16* aTmp,
                       const _Float16* W1, const _Float16* W2, int lane) {
  const int col = lane & 15, rbase = (lane < 16) ? 0 : 8;
#pragma unroll
  for (int j = 0; j < 4; ++j) {            // t = silu(x @ W1)
    v8f acc = {};
#pragma unroll
    for (int c = 0; c < 2; ++c)
      acc = wmma_f16(load_a16(aIn + 32 * c, 64, lane),
                     load_bswz(W1, c * 4 + j, lane), acc);
#pragma unroll
    for (int i = 0; i < 8; ++i)
      aTmp[(rbase + i) * 64 + 16 * j + col] = (_Float16)silu(acc[i]);
  }
#pragma unroll
  for (int j = 0; j < 4; ++j) {            // x = (x + silu(t @ W2)) * INV2
    v8f acc = {};
#pragma unroll
    for (int c = 0; c < 2; ++c)
      acc = wmma_f16(load_a16(aTmp + 32 * c, 64, lane),
                     load_bswz(W2, c * 4 + j, lane), acc);
#pragma unroll
    for (int i = 0; i < 8; ++i) {
      int p = (rbase + i) * 64 + 16 * j + col;
      float nx = (xbuf[p] + silu(acc[i])) * INV2f;
      xbuf[p] = nx;
      aIn[p] = (_Float16)nx;
    }
  }
}

// ---- one basis family in stage1: mn = silu((silu(m_st@Wm)*(rbf@Wr)) @ Wc) --
// Wm: 64x64 swz (tile c*4+j) ; Wr: 32x64 swz (tile j) ; Wc: 64x32 swz (c*2+j).
DEV void stage1_family(const _Float16* am, const _Float16* ar, _Float16* at,
                       const _Float16* Wm, const _Float16* Wr,
                       const _Float16* Wc, float* __restrict__ out, int base,
                       int lane) {
  const int col = lane & 15, rbase = (lane < 16) ? 0 : 8;
#pragma unroll
  for (int j = 0; j < 4; ++j) {            // m = silu(m_st@Wm) * (rbf@Wr)
    v8f a1 = {}, a2 = {};
#pragma unroll
    for (int c = 0; c < 2; ++c)
      a1 = wmma_f16(load_a16(am + 32 * c, 64, lane),
                    load_bswz(Wm, c * 4 + j, lane), a1);
    a2 = wmma_f16(load_a16(ar, 32, lane), load_bswz(Wr, j, lane), a2);
#pragma unroll
    for (int i = 0; i < 8; ++i)
      at[(rbase + i) * 64 + 16 * j + col] = (_Float16)(silu(a1[i]) * a2[i]);
  }
#pragma unroll
  for (int j = 0; j < 2; ++j) {            // mn = silu(m @ Wc)  (E,32)
    v8f acc = {};
#pragma unroll
    for (int c = 0; c < 2; ++c)
      acc = wmma_f16(load_a16(at + 32 * c, 64, lane),
                     load_bswz(Wc, c * 2 + j, lane), acc);
#pragma unroll
    for (int i = 0; i < 8; ++i)
      out[(size_t)(base + rbase + i) * 32 + 16 * j + col] = silu(acc[i]);
  }
}

// ============================================================================
// Kernel 1: per edge: x_st = m_st@W_st ; mnq/mnt for quad & trip families
// ============================================================================
__global__ __launch_bounds__(128) void stage1_kernel(
    const float* __restrict__ m_st, const float* __restrict__ rbf,
    const float* W_st, const float* qWm, const float* qWr, const float* qWc,
    const float* tWm, const float* tWr, const float* tWc,
    float* __restrict__ x_st_out, float* __restrict__ mnq_out,
    float* __restrict__ mnt_out, int E) {
  __shared__ _Float16 wSt[64 * 64], wQm[64 * 64], wQr[32 * 64], wQc[64 * 32];
  __shared__ _Float16 wTm[64 * 64], wTr[32 * 64], wTc[64 * 32];
  __shared__ _Float16 aM[4][16 * 64], aR[4][16 * 32], aT[4][16 * 64];
  const int tid = threadIdx.x, nt = blockDim.x;
  stage_w_swz(wSt, W_st, 64, 64, 64, tid, nt);
  stage_w_swz(wQm, qWm, 64, 64, 64, tid, nt);
  stage_w_swz(wQr, qWr, 32, 64, 16, tid, nt);
  stage_w_swz(wQc, qWc, 64, 32, 64, tid, nt);
  stage_w_swz(wTm, tWm, 64, 64, 64, tid, nt);
  stage_w_swz(wTr, tWr, 32, 64, 16, tid, nt);
  stage_w_swz(wTc, tWc, 64, 32, 64, tid, nt);
  __syncthreads();
  const int wave = tid >> 5, lane = tid & 31;
  const int base = (blockIdx.x * 4 + wave) * 16;
  if (base >= E) return;
  _Float16 *am = aM[wave], *ar = aR[wave], *at = aT[wave];
  for (int i = lane; i < 16 * 64; i += 32)
    am[i] = (_Float16)m_st[(size_t)base * 64 + i];
  for (int i = lane; i < 16 * 32; i += 32) {
    int r = i >> 5, c = i & 31;
    ar[i] = (c < 16) ? (_Float16)rbf[(size_t)(base + r) * 16 + c]
                     : (_Float16)0.0f;
  }
  const int col = lane & 15, rbase = (lane < 16) ? 0 : 8;
  // x_st = m_st @ W_st
#pragma unroll
  for (int j = 0; j < 4; ++j) {
    v8f acc = {};
#pragma unroll
    for (int c = 0; c < 2; ++c)
      acc = wmma_f16(load_a16(am + 32 * c, 64, lane),
                     load_bswz(wSt, c * 4 + j, lane), acc);
#pragma unroll
    for (int i = 0; i < 8; ++i)
      x_st_out[(size_t)(base + rbase + i) * 64 + 16 * j + col] = acc[i];
  }
  stage1_family(am, ar, at, wQm, wQr, wQc, mnq_out, base, lane);
  stage1_family(am, ar, at, wTm, wTr, wTc, mnt_out, base, lane);
}

// ============================================================================
// Kernel 2: basis gather + cbf/sbf projections, nb-reduction -> xq, xt (E,32)
// ============================================================================
__global__ __launch_bounds__(256) void basis_kernel(
    const float* __restrict__ mnq, const float* __restrict__ mnt,
    const float* __restrict__ cbf, const float* __restrict__ sbf,
    const int* __restrict__ idx_s, const int* __restrict__ basis1,
    const int* __restrict__ basis2, const float* qWcbf, const float* qWmsbf,
    const float* qWsbf, const float* tWcbf, float* __restrict__ xq_out,
    float* __restrict__ xt_out, int E) {
  __shared__ _Float16 wQc[32 * 32], wQm[32 * 32], wQs[32 * 32], wTc[32 * 32];
  __shared__ _Float16 aC[8][16 * 32], aS[8][16 * 32], aG[8][16 * 32];
  const int tid = threadIdx.x, nt = blockDim.x;
  stage_w_swz(wQc, qWcbf, 32, 32, 16, tid, nt);
  stage_w_swz(wQm, qWmsbf, 32, 32, 32, tid, nt);
  stage_w_swz(wQs, qWsbf, 32, 32, 32, tid, nt);
  stage_w_swz(wTc, tWcbf, 32, 32, 16, tid, nt);
  __syncthreads();
  const int wave = tid >> 5, lane = tid & 31;
  const int base = (blockIdx.x * 8 + wave) * 16;
  if (base >= E) return;
  _Float16 *ac = aC[wave], *as = aS[wave], *ag = aG[wave];
  const int col = lane & 15, rbase = (lane < 16) ? 0 : 8;
  v8f accQ[2] = {{}, {}}, accT[2] = {{}, {}};
  for (int nb = 0; nb < 4; ++nb) {
    for (int i = lane; i < 16 * 32; i += 32) {
      int r = i >> 5, cc = i & 31;
      size_t rowoff = (size_t)(base + r) * 4 + nb;
      ac[i] = (cc < 16) ? (_Float16)cbf[rowoff * 16 + cc] : (_Float16)0.0f;
      as[i] = (_Float16)sbf[rowoff * 32 + cc];
    }
    v8f cq[2], ct[2], sq[2];
#pragma unroll
    for (int j = 0; j < 2; ++j) {
      v8f z = {};
      v16h a = load_a16(ac, 32, lane);
      cq[j] = wmma_f16(a, load_bswz(wQc, j, lane), z);
      ct[j] = wmma_f16(a, load_bswz(wTc, j, lane), z);
      sq[j] = wmma_f16(load_a16(as, 32, lane), load_bswz(wQs, j, lane), z);
    }
    // L2-resident gathers; index loads hoisted, elementwise in C-lane layout
#pragma unroll
    for (int i = 0; i < 8; ++i) {
      int row = base + rbase + i;
      int s  = idx_s[row];
      int b1 = basis1[s * 4 + nb];
      int b2 = basis2[s * 4 + nb];
      const float* q1 = mnq + (size_t)b1 * 32;
      const float* q2 = mnq + (size_t)b2 * 32;
      const float* t1 = mnt + (size_t)b1 * 32;
#pragma unroll
      for (int j = 0; j < 2; ++j) {
        int cc = 16 * j + col;
        float gq = (q1[cc] + q2[cc]) * INV2f;
        ag[(rbase + i) * 32 + cc] = (_Float16)(gq * cq[j][i]);
        accT[j][i] += t1[cc] * ct[j][i];   // trip path
      }
    }
#pragma unroll
    for (int j = 0; j < 2; ++j) {          // quad: silu(ag@Wmsbf) * sbfproj
      v8f z = {};
      v8f mm = wmma_f16(load_a16(ag, 32, lane), load_bswz(wQm, j, lane), z);
#pragma unroll
      for (int i = 0; i < 8; ++i) accQ[j][i] += silu(mm[i]) * sq[j][i];
    }
  }
#pragma unroll
  for (int j = 0; j < 2; ++j)
#pragma unroll
    for (int i = 0; i < 8; ++i) {
      size_t p = (size_t)(base + rbase + i) * 32 + 16 * j + col;
      xq_out[p] = accQ[j][i] * INVNBf;
      xt_out[p] = accT[j][i] * INVNBf;
    }
}

// ============================================================================
// Kernel 3: x -> silu(x@Wdir) -> silu(@Wst)/silu(@Wts), fold into pre_st/ts
// ============================================================================
__global__ __launch_bounds__(256) void dir_kernel(
    const float* __restrict__ xin, const float* Wdir_g, const float* Wst_g,
    const float* Wts_g, const float* __restrict__ xst_in,
    float* __restrict__ out_st, float* __restrict__ out_ts, int accumulate,
    int E) {
  __shared__ _Float16 wD[32 * 64], wS[64 * 64], wT[64 * 64];
  __shared__ _Float16 aX[8][16 * 32], aD[8][16 * 64];
  const int tid = threadIdx.x, nt = blockDim.x;
  stage_w_swz(wD, Wdir_g, 32, 64, 32, tid, nt);
  stage_w_swz(wS, Wst_g, 64, 64, 64, tid, nt);
  stage_w_swz(wT, Wts_g, 64, 64, 64, tid, nt);
  __syncthreads();
  const int wave = tid >> 5, lane = tid & 31;
  const int base = (blockIdx.x * 8 + wave) * 16;
  if (base >= E) return;
  _Float16 *ax = aX[wave], *ad = aD[wave];
  for (int i = lane; i < 16 * 32; i += 32)
    ax[i] = (_Float16)xin[(size_t)base * 32 + i];
  const int col = lane & 15, rbase = (lane < 16) ? 0 : 8;
#pragma unroll
  for (int j = 0; j < 4; ++j) {            // xdir = silu(x @ Wdir), K=32
    v8f acc = {};
    acc = wmma_f16(load_a16(ax, 32, lane), load_bswz(wD, j, lane), acc);
#pragma unroll
    for (int i = 0; i < 8; ++i)
      ad[(rbase + i) * 64 + 16 * j + col] = (_Float16)silu(acc[i]);
  }
#pragma unroll
  for (int j = 0; j < 4; ++j) {
    v8f as_ = {}, at_ = {};
#pragma unroll
    for (int c = 0; c < 2; ++c) {
      v16h a = load_a16(ad + 32 * c, 64, lane);
      as_ = wmma_f16(a, load_bswz(wS, c * 4 + j, lane), as_);
      at_ = wmma_f16(a, load_bswz(wT, c * 4 + j, lane), at_);
    }
#pragma unroll
    for (int i = 0; i < 8; ++i) {
      size_t p = (size_t)(base + rbase + i) * 64 + 16 * j + col;
      float vs = INV2f * silu(as_[i]);
      float vt = INV2f * silu(at_[i]);
      if (accumulate) { out_st[p] += vs; out_ts[p] += vt; }
      else            { out_st[p] = xst_in[p] + vs; out_ts[p] = vt; }
    }
  }
}

// ============================================================================
// Kernel 4: idx_swap combine, rb/ra residuals -> m_mid; scatter-add atoms
// ============================================================================
__global__ __launch_bounds__(64) void mix_kernel(
    const float* __restrict__ pre_st, const float* __restrict__ ts_buf,
    const int* __restrict__ idx_swap, const float* __restrict__ m_st,
    const float* __restrict__ rbf, const int* __restrict__ idx_t,
    const float* rbW1g, const float* rbW2g, const float* raW1g,
    const float* raW2g, const float* aeWrbf_g, float* __restrict__ m_mid,
    float* __restrict__ xa_atoms, int E) {
  __shared__ _Float16 w1[64 * 64], w2[64 * 64], w3[64 * 64], w4[64 * 64],
      wR[32 * 64];
  __shared__ float xb[2][16 * 64];
  __shared__ _Float16 atl[2][16 * 64], ttl[2][16 * 64], arl[2][16 * 32];
  const int tid = threadIdx.x, nt = blockDim.x;
  stage_w_swz(w1, rbW1g, 64, 64, 64, tid, nt);
  stage_w_swz(w2, rbW2g, 64, 64, 64, tid, nt);
  stage_w_swz(w3, raW1g, 64, 64, 64, tid, nt);
  stage_w_swz(w4, raW2g, 64, 64, 64, tid, nt);
  stage_w_swz(wR, aeWrbf_g, 32, 64, 16, tid, nt);
  __syncthreads();
  const int wave = tid >> 5, lane = tid & 31;
  const int base = (blockIdx.x * 2 + wave) * 16;
  if (base >= E) return;
  float* xbuf = xb[wave];
  _Float16 *atile = atl[wave], *ttile = ttl[wave], *ar = arl[wave];
  for (int i = lane; i < 16 * 64; i += 32) {
    int r = i >> 6, c = i & 63;
    float x = (pre_st[(size_t)(base + r) * 64 + c] +
               ts_buf[(size_t)idx_swap[base + r] * 64 + c]) * INV3f;
    xbuf[i] = x;
    atile[i] = (_Float16)x;
  }
  residual_tile(xbuf, atile, ttile, w1, w2, lane);
  for (int i = lane; i < 16 * 64; i += 32) {
    float x = (m_st[(size_t)base * 64 + i] + xbuf[i]) * INV2f;
    xbuf[i] = x;
    atile[i] = (_Float16)x;
  }
  residual_tile(xbuf, atile, ttile, w3, w4, lane);
  for (int i = lane; i < 16 * 64; i += 32)
    m_mid[(size_t)base * 64 + i] = xbuf[i];
  for (int i = lane; i < 16 * 32; i += 32) {
    int r = i >> 5, c = i & 31;
    ar[i] = (c < 16) ? (_Float16)rbf[(size_t)(base + r) * 16 + c]
                     : (_Float16)0.0f;
  }
  const int col = lane & 15, rbase = (lane < 16) ? 0 : 8;
#pragma unroll
  for (int j = 0; j < 4; ++j) {            // xa_e = m * (rbf @ ae_Wrbf)
    v8f acc = {};
    acc = wmma_f16(load_a16(ar, 32, lane), load_bswz(wR, j, lane), acc);
#pragma unroll
    for (int i = 0; i < 8; ++i) {
      int row = base + rbase + i;
      float v = xbuf[(rbase + i) * 64 + 16 * j + col] * acc[i];
      atomicAdd(&xa_atoms[(size_t)idx_t[row] * 64 + 16 * j + col], v);
    }
  }
}

// ============================================================================
// Kernel 5: per atom: silu(xa@Wdense) -> ae residual -> h_new
// ============================================================================
__global__ __launch_bounds__(64) void atom_kernel(
    const float* __restrict__ xa_atoms, const float* __restrict__ h_in,
    const float* Wd_g, const float* W1g, const float* W2g,
    float* __restrict__ h_out, int N) {
  __shared__ _Float16 wD[64 * 64], w1[64 * 64], w2[64 * 64];
  __shared__ float xb[2][16 * 64];
  __shared__ _Float16 atl[2][16 * 64], ttl[2][16 * 64];
  const int tid = threadIdx.x, nt = blockDim.x;
  stage_w_swz(wD, Wd_g, 64, 64, 64, tid, nt);
  stage_w_swz(w1, W1g, 64, 64, 64, tid, nt);
  stage_w_swz(w2, W2g, 64, 64, 64, tid, nt);
  __syncthreads();
  const int wave = tid >> 5, lane = tid & 31;
  const int base = (blockIdx.x * 2 + wave) * 16;
  if (base >= N) return;
  float* xbuf = xb[wave];
  _Float16 *atile = atl[wave], *ttile = ttl[wave];
  for (int i = lane; i < 16 * 64; i += 32)
    atile[i] = (_Float16)xa_atoms[(size_t)base * 64 + i];
  const int col = lane & 15, rbase = (lane < 16) ? 0 : 8;
#pragma unroll
  for (int j = 0; j < 4; ++j) {            // silu(xa @ ae_Wdense)
    v8f acc = {};
#pragma unroll
    for (int c = 0; c < 2; ++c)
      acc = wmma_f16(load_a16(atile + 32 * c, 64, lane),
                     load_bswz(wD, c * 4 + j, lane), acc);
#pragma unroll
    for (int i = 0; i < 8; ++i) {
      int p = (rbase + i) * 64 + 16 * j + col;
      float v = silu(acc[i]);
      xbuf[p] = v;
      ttile[p] = (_Float16)v;
    }
  }
  residual_tile(xbuf, ttile, atile, w1, w2, lane);
  for (int i = lane; i < 16 * 64; i += 32)
    h_out[(size_t)base * 64 + i] =
        (h_in[(size_t)base * 64 + i] + xbuf[i]) * INV2f;
}

// ============================================================================
// Kernel 6: asi concat(192) GEMM + ras residual -> m_new
// ============================================================================
__global__ __launch_bounds__(64) void asi_kernel(
    const float* __restrict__ h_new, const float* __restrict__ m_mid,
    const int* __restrict__ idx_s, const int* __restrict__ idx_t,
    const float* asiW_g, const float* W1g, const float* W2g,
    float* __restrict__ m_out, int E) {
  __shared__ _Float16 wA[192 * 64], w1[64 * 64], w2[64 * 64];
  __shared__ float xb[2][16 * 64];
  __shared__ _Float16 cst_[2][16 * 32], atl[2][16 * 64], ttl[2][16 * 64];
  const int tid = threadIdx.x, nt = blockDim.x;
  stage_w_swz(wA, asiW_g, 192, 64, 192, tid, nt);
  stage_w_swz(w1, W1g, 64, 64, 64, tid, nt);
  stage_w_swz(w2, W2g, 64, 64, 64, tid, nt);
  __syncthreads();
  const int wave = tid >> 5, lane = tid & 31;
  const int base = (blockIdx.x * 2 + wave) * 16;
  if (base >= E) return;
  float* xbuf = xb[wave];
  _Float16 *cst = cst_[wave], *atile = atl[wave], *ttile = ttl[wave];
  const int col = lane & 15, rbase = (lane < 16) ? 0 : 8;
  v8f acc[4] = {{}, {}, {}, {}};
  for (int c = 0; c < 6; ++c) {            // K = 192 in six 32-chunks
    for (int i = lane; i < 16 * 32; i += 32) {
      int r = i >> 5, cc = i & 31;
      int gcol = c * 32 + cc;
      float v;
      if (gcol < 64)       v = h_new[(size_t)idx_s[base + r] * 64 + gcol];
      else if (gcol < 128) v = h_new[(size_t)idx_t[base + r] * 64 + gcol - 64];
      else                 v = m_mid[(size_t)(base + r) * 64 + gcol - 128];
      cst[i] = (_Float16)v;
    }
    v16h a = load_a16(cst, 32, lane);
#pragma unroll
    for (int j = 0; j < 4; ++j)
      acc[j] = wmma_f16(a, load_bswz(wA, c * 4 + j, lane), acc[j]);
  }
#pragma unroll
  for (int j = 0; j < 4; ++j)
#pragma unroll
    for (int i = 0; i < 8; ++i) {
      int p = (rbase + i) * 64 + 16 * j + col;
      float v = silu(acc[j][i]);
      xbuf[p] = v;
      atile[p] = (_Float16)v;
    }
  residual_tile(xbuf, atile, ttile, w1, w2, lane);
  for (int i = lane; i < 16 * 64; i += 32)
    m_out[(size_t)base * 64 + i] =
        (m_mid[(size_t)base * 64 + i] + xbuf[i]) * INV2f;
}

// ============================================================================
extern "C" void kernel_launch(void* const* d_in, const int* in_sizes, int n_in,
                              void* d_out, int out_size, void* d_ws,
                              size_t ws_size, hipStream_t stream) {
  const float* h      = (const float*)d_in[0];
  const float* m_st   = (const float*)d_in[1];
  const float* rbf    = (const float*)d_in[2];
  const float* cbf    = (const float*)d_in[3];
  const float* sbf    = (const float*)d_in[4];
  const int* idx_s    = (const int*)d_in[5];
  const int* idx_t    = (const int*)d_in[6];
  const int* idx_swap = (const int*)d_in[7];
  const int* basis1   = (const int*)d_in[8];
  const int* basis2   = (const int*)d_in[9];
  const float* W_st    = (const float*)d_in[10];
  const float* qWmrbf  = (const float*)d_in[11];
  const float* qWrbf   = (const float*)d_in[12];
  const float* qWmcbf  = (const float*)d_in[13];
  const float* qWcbf   = (const float*)d_in[14];
  const float* qWmsbf  = (const float*)d_in[15];
  const float* qWsbf   = (const float*)d_in[16];
  const float* qWdir   = (const float*)d_in[17];
  const float* qWst    = (const float*)d_in[18];
  const float* qWts    = (const float*)d_in[19];
  const float* tWmrbf  = (const float*)d_in[20];
  const float* tWrbf   = (const float*)d_in[21];
  const float* tWmcbf  = (const float*)d_in[22];
  const float* tWcbf   = (const float*)d_in[23];
  const float* tWdir   = (const float*)d_in[24];
  const float* tWst    = (const float*)d_in[25];
  const float* tWts    = (const float*)d_in[26];
  const float* rbW1    = (const float*)d_in[27];
  const float* rbW2    = (const float*)d_in[28];
  const float* raW1    = (const float*)d_in[29];
  const float* raW2    = (const float*)d_in[30];
  const float* aeWrbf  = (const float*)d_in[31];
  const float* aeWdense= (const float*)d_in[32];
  const float* aeW1    = (const float*)d_in[33];
  const float* aeW2    = (const float*)d_in[34];
  const float* asiW    = (const float*)d_in[35];
  const float* rasW1   = (const float*)d_in[36];
  const float* rasW2   = (const float*)d_in[37];

  const int N = in_sizes[0] / 64;
  const int E = in_sizes[1] / 64;

  float* ws    = (float*)d_ws;
  float* x_st  = ws;                         // E*64 (later: pre_st, in-place)
  float* mnq   = x_st + (size_t)E * 64;      // E*32
  float* mnt   = mnq + (size_t)E * 32;       // E*32
  float* xq    = mnt + (size_t)E * 32;       // E*32
  float* xt    = xq + (size_t)E * 32;        // E*32
  float* tsbuf = xt + (size_t)E * 32;        // E*64
  float* m_mid = tsbuf + (size_t)E * 64;     // E*64
  float* xa    = m_mid + (size_t)E * 64;     // N*64
  float* h_new = (float*)d_out;              // N*64
  float* m_new = h_new + (size_t)N * 64;     // E*64

  (void)hipMemsetAsync(xa, 0, (size_t)N * 64 * sizeof(float), stream);

  const int tilesE = (E + 15) / 16;
  const int tilesN = (N + 15) / 16;

  stage1_kernel<<<(tilesE + 3) / 4, 128, 0, stream>>>(
      m_st, rbf, W_st, qWmrbf, qWrbf, qWmcbf, tWmrbf, tWrbf, tWmcbf, x_st, mnq,
      mnt, E);
  basis_kernel<<<(tilesE + 7) / 8, 256, 0, stream>>>(
      mnq, mnt, cbf, sbf, idx_s, basis1, basis2, qWcbf, qWmsbf, qWsbf, tWcbf,
      xq, xt, E);
  dir_kernel<<<(tilesE + 7) / 8, 256, 0, stream>>>(
      xq, qWdir, qWst, qWts, x_st, x_st, tsbuf, /*accumulate=*/0, E);
  dir_kernel<<<(tilesE + 7) / 8, 256, 0, stream>>>(
      xt, tWdir, tWst, tWts, x_st, x_st, tsbuf, /*accumulate=*/1, E);
  mix_kernel<<<(tilesE + 1) / 2, 64, 0, stream>>>(
      x_st, tsbuf, idx_swap, m_st, rbf, idx_t, rbW1, rbW2, raW1, raW2, aeWrbf,
      m_mid, xa, E);
  atom_kernel<<<(tilesN + 1) / 2, 64, 0, stream>>>(xa, h, aeWdense, aeW1, aeW2,
                                                   h_new, N);
  asi_kernel<<<(tilesE + 1) / 2, 64, 0, stream>>>(h_new, m_mid, idx_s, idx_t,
                                                  asiW, rasW1, rasW2, m_new, E);
}
